// ChebNet_24189255811803
// MI455X (gfx1250) — matrix operand order repaired and checked by reading, not compiled, not run on hardware
//
#include <hip/hip_runtime.h>
#include <hip/hip_bf16.h>

// ---------------- problem constants (match reference) ----------------
#define NN   50000      // nodes
#define NE   800000     // edges
#define FH   128        // F_IN == H == 128
#define NCLS 10
#define NB   8

typedef __attribute__((ext_vector_type(16))) __bf16 v16bf;
typedef __attribute__((ext_vector_type(8)))  float  v8f;

// padded LDS row stride (bf16 elems) to stagger banks
#define LDSK 136

// =====================================================================
// GEMM: Y [nrows,128] (+)= X [nrows,128] @ W [128,128]   (bf16 WMMA, f32 acc)
// block: 256 thr = 8 waves; block tile 64 rows x 128 cols;
// wave w owns cols [16w,16w+16), row-tiles t=0..3; K loop 4 x 32.
// =====================================================================
__global__ __launch_bounds__(256) void gemm128_wmma(
    const float* __restrict__ X, const float* __restrict__ W,
    float* __restrict__ Y, int nrows, int accumulate)
{
    __shared__ __bf16 sX[64 * LDSK];     // [m][k]
    __shared__ __bf16 sW[128 * LDSK];    // transposed: [n][k]

    const int tid = threadIdx.x;
    const int rowbase = blockIdx.x * 64;

    // W -> LDS (transposed, f32->bf16). Global read coalesced along n.
    for (int idx = tid; idx < 128 * 128; idx += 256) {
        int k = idx >> 7, n = idx & 127;
        sW[n * LDSK + k] = (__bf16)W[idx];
    }
    // X tile -> LDS (f32->bf16)
    for (int idx = tid; idx < 64 * 128; idx += 256) {
        int m = idx >> 7, k = idx & 127;
        int r = rowbase + m;
        float v = (r < nrows) ? X[r * FH + k] : 0.0f;
        sX[m * LDSK + k] = (__bf16)v;
    }
    __syncthreads();

    const int lane = tid & 31;
    const int wave = tid >> 5;
    const int r    = lane & 15;
    const int h    = lane >> 4;       // lane half-group
    const int n0   = wave * 16;

    v8f acc[4];
    if (accumulate) {
        #pragma unroll
        for (int t = 0; t < 4; ++t) {
            const int mb = rowbase + t * 16 + h * 8;   // C layout: VGPR j -> M=j(+8)
            const int n  = n0 + r;                     //           lane  -> N
            #pragma unroll
            for (int j = 0; j < 8; ++j) {
                int row = mb + j;
                acc[t][j] = (row < nrows) ? Y[row * FH + n] : 0.0f;
            }
        }
    } else {
        #pragma unroll
        for (int t = 0; t < 4; ++t)
            #pragma unroll
            for (int j = 0; j < 8; ++j) acc[t][j] = 0.0f;
    }

    #pragma unroll
    for (int kk = 0; kk < 4; ++kk) {
        const int kbase = kk * 32;
        // B fragment (32x16): lane group h holds K = h*16 .. h*16+15 (contiguous)
        v16bf b;
        {
            const __bf16* pb = &sW[(n0 + r) * LDSK + kbase + h * 16];
            #pragma unroll
            for (int i = 0; i < 16; ++i) b[i] = pb[i];
        }
        #pragma unroll
        for (int t = 0; t < 4; ++t) {
            // A fragment (16x32): lane group h holds K = h*16 .. h*16+15 (contiguous)
            v16bf a;
            const __bf16* pa = &sX[(t * 16 + r) * LDSK + kbase + h * 16];
            #pragma unroll
            for (int i = 0; i < 16; ++i) a[i] = pa[i];
            acc[t] = __builtin_amdgcn_wmma_f32_16x16x32_bf16(
                false, a, false, b, (short)0, acc[t], false, false);
        }
    }

    #pragma unroll
    for (int t = 0; t < 4; ++t) {
        const int mb = rowbase + t * 16 + h * 8;
        const int n  = n0 + r;
        #pragma unroll
        for (int j = 0; j < 8; ++j) {
            int row = mb + j;
            if (row < nrows) Y[row * FH + n] = acc[t][j];
        }
    }
}

// =====================================================================
// graph / elementwise kernels
// =====================================================================
__global__ void fill_f32(float* p, float v, int n) {
    int i = blockIdx.x * blockDim.x + threadIdx.x;
    if (i < n) p[i] = v;
}

__global__ void degree_kernel(const long long* __restrict__ ei, float* deg, int E) {
    int e = blockIdx.x * blockDim.x + threadIdx.x;
    if (e < E) atomicAdd(&deg[(int)ei[e]], 1.0f);
}

__global__ void node_prep(const float* __restrict__ deg,
                          const long long* __restrict__ batch,
                          const float* __restrict__ lam,
                          float* __restrict__ dis, float* __restrict__ diag, int n) {
    int i = blockIdx.x * blockDim.x + threadIdx.x;
    if (i >= n) return;
    float d = deg[i];
    dis[i]  = (d > 0.0f) ? rsqrtf(fmaxf(d, 1e-12f)) : 0.0f;
    diag[i] = 2.0f / lam[(int)batch[i]] - 1.0f;
}

__global__ void edge_weight(const long long* __restrict__ ei,
                            const long long* __restrict__ batch,
                            const float* __restrict__ lam,
                            const float* __restrict__ dis,
                            float* __restrict__ we, int E) {
    int e = blockIdx.x * blockDim.x + threadIdx.x;
    if (e >= E) return;
    int r = (int)ei[e], c = (int)ei[E + e];
    we[e] = -2.0f * dis[r] * dis[c] / lam[(int)batch[r]];
}

// Y = scale*diag[node]*V  (- sub, if given);  i over N*128
__global__ void lhat_init(const float* __restrict__ V, const float* __restrict__ diag,
                          const float* __restrict__ sub, float* __restrict__ Y,
                          float scale, int n) {
    int i = blockIdx.x * blockDim.x + threadIdx.x;
    if (i >= n) return;
    float y = scale * diag[i >> 7] * V[i];
    if (sub) y -= sub[i];
    Y[i] = y;
}

// Y[row[e]] += scale*we[e]*V[col[e]]; 32 threads/edge, 4 feats each
__global__ void lhat_scatter(const long long* __restrict__ ei,
                             const float* __restrict__ we,
                             const float* __restrict__ V,
                             float* __restrict__ Y, float scale, int E) {
    int t = blockIdx.x * blockDim.x + threadIdx.x;
    int e = t >> 5;
    if (e >= E) return;
    int fc = (t & 31) * 4;
    int r = (int)ei[e], c = (int)ei[E + e];
    float w = scale * we[e];
    const float4 v = *(const float4*)&V[c * FH + fc];
    float* y = &Y[r * FH + fc];
    atomicAdd(y + 0, w * v.x);
    atomicAdd(y + 1, w * v.y);
    atomicAdd(y + 2, w * v.z);
    atomicAdd(y + 3, w * v.w);
}

__global__ void bias_relu(const float* __restrict__ A, const float* __restrict__ b,
                          float* __restrict__ O, int n) {
    int i = blockIdx.x * blockDim.x + threadIdx.x;
    if (i >= n) return;
    float v = A[i] + b[i & (FH - 1)];
    O[i] = v > 0.0f ? v : 0.0f;
}

// mean-pool partial sums: LDS [B][128] per block, then global atomics
__global__ __launch_bounds__(256) void pool_kernel(const float* __restrict__ Hf,
                                                   const long long* __restrict__ batch,
                                                   float* pooled, float* counts, int n) {
    __shared__ float sp[NB * FH];
    __shared__ float sc[NB];
    int tid = threadIdx.x;
    for (int i = tid; i < NB * FH; i += 256) sp[i] = 0.0f;
    if (tid < NB) sc[tid] = 0.0f;
    __syncthreads();
    const int start = blockIdx.x * 128;       // 128 nodes per block
    const int f = tid & 127, half = tid >> 7; // 2 node-streams
    for (int m = half; m < 128; m += 2) {
        int node = start + m;
        if (node < n) {
            int b = (int)batch[node];
            atomicAdd(&sp[b * FH + f], Hf[node * FH + f]);
            if (f == 0) atomicAdd(&sc[b], 1.0f);
        }
    }
    __syncthreads();
    for (int i = tid; i < NB * FH; i += 256) atomicAdd(&pooled[i], sp[i]);
    if (tid < NB) atomicAdd(&counts[tid], sc[tid]);
}

__global__ void final_kernel(const float* __restrict__ pooled,
                             const float* __restrict__ counts,
                             const float* __restrict__ Wl,
                             const float* __restrict__ bl, float* out) {
    int t = threadIdx.x;
    if (t >= NB * NCLS) return;
    int b = t / NCLS, c = t % NCLS;
    float inv = 1.0f / fmaxf(counts[b], 1.0f);
    float s = 0.0f;
    for (int k = 0; k < FH; ++k) s += pooled[b * FH + k] * Wl[k * NCLS + c];
    out[t] = s * inv + bl[c];
}

// =====================================================================
// host-side orchestration
// =====================================================================
static void run_cheb_layer(const float* X, const float* Wstack, const float* bias,
                           const long long* ei, const float* we, const float* diag,
                           float* TA, float* TB, float* TC, float* ACC, float* OUT,
                           hipStream_t s)
{
    const int gg   = (NN + 63) / 64;                 // gemm grid
    const int gnh  = (NN * FH + 255) / 256;          // elementwise grid
    const int gsc  = (NE * 32 + 255) / 256;          // scatter grid
    const int WSZ  = FH * FH;

    // k=0
    gemm128_wmma<<<gg, 256, 0, s>>>(X, Wstack + 0 * WSZ, ACC, NN, 0);
    // k=1: TA = Lhat(X)
    lhat_init<<<gnh, 256, 0, s>>>(X, diag, nullptr, TA, 1.0f, NN * FH);
    lhat_scatter<<<gsc, 256, 0, s>>>(ei, we, X, TA, 1.0f, NE);
    gemm128_wmma<<<gg, 256, 0, s>>>(TA, Wstack + 1 * WSZ, ACC, NN, 1);
    // k=2: TB = 2*Lhat(TA) - X
    lhat_init<<<gnh, 256, 0, s>>>(TA, diag, X, TB, 2.0f, NN * FH);
    lhat_scatter<<<gsc, 256, 0, s>>>(ei, we, TA, TB, 2.0f, NE);
    gemm128_wmma<<<gg, 256, 0, s>>>(TB, Wstack + 2 * WSZ, ACC, NN, 1);
    // k=3: TC = 2*Lhat(TB) - TA
    lhat_init<<<gnh, 256, 0, s>>>(TB, diag, TA, TC, 2.0f, NN * FH);
    lhat_scatter<<<gsc, 256, 0, s>>>(ei, we, TB, TC, 2.0f, NE);
    gemm128_wmma<<<gg, 256, 0, s>>>(TC, Wstack + 3 * WSZ, ACC, NN, 1);
    // k=4: TA = 2*Lhat(TC) - TB   (TA's old contents no longer needed)
    lhat_init<<<gnh, 256, 0, s>>>(TC, diag, TB, TA, 2.0f, NN * FH);
    lhat_scatter<<<gsc, 256, 0, s>>>(ei, we, TC, TA, 2.0f, NE);
    gemm128_wmma<<<gg, 256, 0, s>>>(TA, Wstack + 4 * WSZ, ACC, NN, 1);
    // OUT = relu(ACC + bias)   (OUT may alias TB: last read of TB was above)
    bias_relu<<<gnh, 256, 0, s>>>(ACC, bias, OUT, NN * FH);
}

extern "C" void kernel_launch(void* const* d_in, const int* in_sizes, int n_in,
                              void* d_out, int out_size, void* d_ws, size_t ws_size,
                              hipStream_t stream) {
    (void)in_sizes; (void)n_in; (void)out_size; (void)ws_size;

    const float*     x     = (const float*)d_in[0];
    const long long* ei    = (const long long*)d_in[1];   // [2,E] int64
    const long long* batch = (const long long*)d_in[2];   // [N] int64
    const float*     lam   = (const float*)d_in[3];
    const float*     W1    = (const float*)d_in[4];
    const float*     b1    = (const float*)d_in[5];
    const float*     W2    = (const float*)d_in[6];
    const float*     b2    = (const float*)d_in[7];
    const float*     Wlin  = (const float*)d_in[8];
    const float*     blin  = (const float*)d_in[9];
    float*           out   = (float*)d_out;

    // workspace carve-up (256B aligned)
    char* w = (char*)d_ws;
    size_t off = 0;
    auto take = [&](size_t nfloats) {
        float* p = (float*)(w + off);
        off += ((nfloats * 4 + 255) / 256) * 256;
        return p;
    };
    float* deg    = take(NN);
    float* dis    = take(NN);
    float* diag   = take(NN);
    float* we     = take(NE);
    float* TA     = take((size_t)NN * FH);
    float* TB     = take((size_t)NN * FH);
    float* TC     = take((size_t)NN * FH);
    float* ACC    = take((size_t)NN * FH);
    float* H1     = take((size_t)NN * FH);
    float* pooled = take(NB * FH);
    float* counts = take(NB);

    // ---- graph prep ----
    fill_f32<<<(NN + 255) / 256, 256, 0, stream>>>(deg, 0.0f, NN);
    degree_kernel<<<(NE + 255) / 256, 256, 0, stream>>>(ei, deg, NE);
    node_prep<<<(NN + 255) / 256, 256, 0, stream>>>(deg, batch, lam, dis, diag, NN);
    edge_weight<<<(NE + 255) / 256, 256, 0, stream>>>(ei, batch, lam, dis, we, NE);

    // ---- layer 1: x -> H1 ----
    run_cheb_layer(x, W1, b1, ei, we, diag, TA, TB, TC, ACC, H1, stream);
    // ---- layer 2: H1 -> TB (reused as H2) ----
    run_cheb_layer(H1, W2, b2, ei, we, diag, TA, TB, TC, ACC, TB, stream);

    // ---- mean pool + head ----
    fill_f32<<<(NB * FH + 255) / 256, 256, 0, stream>>>(pooled, 0.0f, NB * FH);
    fill_f32<<<1, 256, 0, stream>>>(counts, 0.0f, NB);
    pool_kernel<<<(NN + 127) / 128, 256, 0, stream>>>(TB, batch, pooled, counts, NN);
    final_kernel<<<1, 128, 0, stream>>>(pooled, counts, Wlin, blin, out);
}
